// Contrast_GNNmodel_58342835749568
// MI455X (gfx1250) — matrix operand order repaired and checked by reading, not compiled, not run on hardware
//
#include <hip/hip_runtime.h>
#include <hip/hip_bf16.h>
#include <stdint.h>

// ---------------------------------------------------------------------------
// Types for CDNA5 WMMA (wave32): V_WMMA_F32_16X16X32_BF16
// ---------------------------------------------------------------------------
typedef __attribute__((ext_vector_type(16))) __bf16 bf16x16;
typedef __attribute__((ext_vector_type(8)))  __bf16 bf16x8;
typedef __attribute__((ext_vector_type(8)))  float  f32x8;

union FragB16 {
    bf16x16 v;
    bf16x8  h[2];
    unsigned short u[16];
};

#define IMG_W 64   // image width == feature dim of the [N,64] node matrix
#define CHN   64   // conv channels (HID)
#define PW    66   // padded row width (1-pixel zero halo on each side)

// ---------------------------------------------------------------------------
// Small utility kernels (f32 VALU path — precision-critical / tiny work)
// ---------------------------------------------------------------------------
__global__ __launch_bounds__(256) void fill_zero_f32(float* p, long long n) {
    long long i = (long long)blockIdx.x * blockDim.x + threadIdx.x;
    if (i < n) p[i] = 0.0f;
}

// Zero the halo pixels (all 64 channels) of a padded bf16 activation buffer.
// Padded dims: (H+2) rows x 66 cols; halo = top row, bottom row, col 0, col 65.
__global__ __launch_bounds__(256) void pad_zero_kernel(__bf16* buf, int H) {
    long long nPad = 132 + 2LL * H;
    long long t = (long long)blockIdx.x * blockDim.x + threadIdx.x;
    if (t >= nPad) return;
    size_t pp;
    if (t < 66)       pp = (size_t)t;                              // top halo row
    else if (t < 132) pp = (size_t)(H + 1) * PW + (t - 66);        // bottom halo row
    else {
        long long i = t - 132;
        long long r = i >> 1;
        pp = (size_t)(r + 1) * PW + ((i & 1) ? 65 : 0);            // side halos
    }
    uint4* p = (uint4*)(buf + pp * CHN);   // 64 ch * 2B = 128B = 8 x uint4
#pragma unroll
    for (int k = 0; k < 8; ++k) p[k] = make_uint4(0, 0, 0, 0);
}

// deg[dst] += 1 over all edges (edge_index is int64 [2,E])
__global__ __launch_bounds__(256) void degree_kernel(const long long* ei, long long E, float* deg) {
    long long e = (long long)blockIdx.x * blockDim.x + threadIdx.x;
    if (e >= E) return;
    long long d = ei[E + e];
    atomicAdd(deg + d, 1.0f);
}

__global__ __launch_bounds__(256) void rsqrt_kernel(float* deg, int n) {
    int i = blockIdx.x * blockDim.x + threadIdx.x;
    if (i < n) deg[i] = rsqrtf(deg[i] + 1.0f);   // in-place: deg -> D^-1/2
}

// out[n,j] = sum_k (A[n,k] (+A2[n,k])) * W[k,j] (+ bias[j])
__global__ __launch_bounds__(256) void gemm_small_kernel(
    const float* __restrict__ A, const float* __restrict__ A2,
    const float* __restrict__ W, const float* __restrict__ bias,
    float* __restrict__ out, int n, int K, int F)
{
    long long idx = (long long)blockIdx.x * blockDim.x + threadIdx.x;
    if (idx >= (long long)n * F) return;
    int row = (int)(idx / F), j = (int)(idx % F);
    const float* ar  = A + (size_t)row * K;
    const float* ar2 = A2 ? A2 + (size_t)row * K : nullptr;
    float acc = bias ? bias[j] : 0.0f;
    for (int k = 0; k < K; ++k) {
        float a = ar[k];
        if (ar2) a += ar2[k];
        acc += a * W[(size_t)k * F + j];
    }
    out[idx] = acc;
}

// agg[dst] += xw[src] * dis[src]*dis[dst]  (F is multiple of 4)
__global__ __launch_bounds__(256) void scatter_kernel(
    const float* __restrict__ xw, const float* __restrict__ dis,
    const long long* __restrict__ ei, long long E,
    float* __restrict__ agg, int F)
{
    long long e = (long long)blockIdx.x * blockDim.x + threadIdx.x;
    if (e >= E) return;
    long long s = ei[e], d = ei[E + e];
    float coef = dis[s] * dis[d];
    const float4* xr = (const float4*)(xw + (size_t)s * F);
    float* ar = agg + (size_t)d * F;
    for (int i = 0; i < F / 4; ++i) {
        float4 v = xr[i];
        atomicAdd(ar + 4 * i + 0, v.x * coef);
        atomicAdd(ar + 4 * i + 1, v.y * coef);
        atomicAdd(ar + 4 * i + 2, v.z * coef);
        atomicAdd(ar + 4 * i + 3, v.w * coef);
    }
}

// out = agg + xw * dis^2 + b   (GCN self-loop term + bias)
__global__ __launch_bounds__(256) void gcn_finalize_kernel(
    float* __restrict__ out, const float* __restrict__ agg,
    const float* __restrict__ xw, const float* __restrict__ dis,
    const float* __restrict__ b, int n, int F)
{
    long long idx = (long long)blockIdx.x * blockDim.x + threadIdx.x;
    if (idx >= (long long)n * F) return;
    int row = (int)(idx / F), f = (int)(idx % F);
    float di = dis[row];
    out[idx] = agg[idx] + xw[idx] * di * di + b[f];
}

// Convert conv weights [co][ci][3][3] f32 -> bf16 wbuf[tap][co][ci]
__global__ __launch_bounds__(256) void weight_cvt_kernel(
    const float* __restrict__ W, __bf16* __restrict__ wbuf)
{
    int idx = blockIdx.x * blockDim.x + threadIdx.x;   // co*576 + ci*9 + tap
    if (idx >= 64 * 64 * 9) return;
    int co = idx / 576;
    int rem = idx % 576;
    int ci = rem / 9;
    int tap = rem % 9;
    wbuf[((size_t)tap * 64 + co) * 64 + ci] = (__bf16)W[idx];
}

// ---------------------------------------------------------------------------
// cnn1: C_in = 1 -> 64 channels. h[N,64] f32 image -> padded bf16 NHWC act
// ---------------------------------------------------------------------------
__global__ __launch_bounds__(256) void cnn1_kernel(
    const float* __restrict__ h, const float* __restrict__ w1 /*[64][9]*/,
    const float* __restrict__ b1, __bf16* __restrict__ out, int H)
{
    __shared__ float sw[64 * 9];
    __shared__ float sb[64];
    for (int i = threadIdx.x; i < 576; i += 256) sw[i] = w1[i];
    if (threadIdx.x < 64) sb[threadIdx.x] = b1[threadIdx.x];
    __syncthreads();

    long long p = (long long)blockIdx.x * blockDim.x + threadIdx.x;
    if (p >= (long long)H * IMG_W) return;
    int r = (int)(p >> 6), c = (int)(p & 63);

    float v[9];
#pragma unroll
    for (int t = 0; t < 9; ++t) {
        int rr = r + t / 3 - 1, cc = c + t % 3 - 1;
        v[t] = (rr >= 0 && rr < H && cc >= 0 && cc < IMG_W) ? h[(size_t)rr * IMG_W + cc] : 0.0f;
    }
    __bf16* orow = out + ((size_t)(r + 1) * PW + (c + 1)) * CHN;   // padded layout
    for (int co = 0; co < CHN; ++co) {
        float acc = sb[co];
#pragma unroll
        for (int t = 0; t < 9; ++t) acc += sw[co * 9 + t] * v[t];
        orow[co] = (__bf16)acc;
    }
}

// ---------------------------------------------------------------------------
// WMMA implicit-GEMM 3x3 conv, 64->64 channels, padded bf16 NHWC activations.
// Wave tile: 32 pixels x 32 out-channels = 2x2 WMMA tiles, 4 f32x8 accums.
// Per K-step (9 taps x 2 ci-slices): 2 A-frags (global) + 2 B-frags (LDS)
// feed 4 v_wmma_f32_16x16x32_bf16 -> 72 WMMA / wave, fully branch-free
// (zero halo replaces boundary predication; EXEC stays all-1s throughout).
// Workgroup = 256 thr = 8 waves = 4 col-pairs(2 rows) x 2 co-halves.
// Weights (9*64*64 bf16 = 72KB) staged in dynamic LDS once per workgroup.
// Fragment lane layouts per CDNA5 ISA 7.12.2 (bf16 A 16x32 / B 32x16).
// ---------------------------------------------------------------------------
__global__ __launch_bounds__(256) void conv3x3_wmma_kernel(
    const __bf16* __restrict__ in, const __bf16* __restrict__ wbuf /*[9][64][64]*/,
    const float* __restrict__ bias, const __bf16* __restrict__ residual,
    __bf16* __restrict__ out, int H)
{
    extern __shared__ char smem_raw[];
    __bf16* swb = (__bf16*)smem_raw;
    {   // cooperative stage: 73728 B = 4608 x uint4
        const uint4* srcw = (const uint4*)wbuf;
        uint4* dstw = (uint4*)smem_raw;
        for (int i = threadIdx.x; i < (9 * 64 * 64) / 8; i += 256) dstw[i] = srcw[i];
    }
    __syncthreads();

    const int wave = threadIdx.x >> 5;
    const int lane = threadIdx.x & 31;
    const int half = lane >> 4;
    const int ln   = lane & 15;

    const int pt0    = blockIdx.x * 8 + (wave >> 1) * 2;  // even 16-px tile index
    const int row    = pt0 >> 2;                          // interior image row
    const int c0     = (pt0 & 3) * 16;                    // 0 or 32
    const int cobase = (wave & 1) * 32;                   // co half

    const int kA0 = half * 8;        // A elems 0..7  -> K kA0..kA0+7
    const int kA1 = 16 + half * 8;   // A elems 8..15 -> K kA1..kA1+7
    const int kB  = half * 16;       // unused; B loads use kA0/kA1 (same K order as A)

    // Prefetch the three padded input rows this wave consumes (gfx1250 path).
    __builtin_prefetch(in + ((size_t)(row + 0) * PW + c0 + lane + 1) * CHN, 0, 1);
    __builtin_prefetch(in + ((size_t)(row + 1) * PW + c0 + lane + 1) * CHN, 0, 1);
    __builtin_prefetch(in + ((size_t)(row + 2) * PW + c0 + lane + 1) * CHN, 0, 1);

    f32x8 acc00 = {}, acc01 = {}, acc10 = {}, acc11 = {};

#pragma unroll
    for (int tap = 0; tap < 9; ++tap) {
        const int dr = tap / 3 - 1, dc = tap % 3 - 1;
        // padded pixel: (row+dr+1, c+dc+1) — always in-bounds, halo is zero
        const __bf16* arow0 = in + ((size_t)(row + dr + 1) * PW + (c0 + ln + dc + 1)) * CHN;
        const __bf16* arow1 = arow0 + 16 * CHN;
        const __bf16* wrow0 = swb + ((size_t)tap * 64 + cobase + ln) * 64;
        const __bf16* wrow1 = wrow0 + 16 * 64;
#pragma unroll
        for (int cib = 0; cib < CHN; cib += 32) {
            FragB16 a0, a1, b0, b1;
            a0.h[0] = *(const bf16x8*)(arow0 + cib + kA0);
            a0.h[1] = *(const bf16x8*)(arow0 + cib + kA1);
            a1.h[0] = *(const bf16x8*)(arow1 + cib + kA0);
            a1.h[1] = *(const bf16x8*)(arow1 + cib + kA1);
            b0.h[0] = *(const bf16x8*)(wrow0 + cib + kA0);
            b0.h[1] = *(const bf16x8*)(wrow0 + cib + kA1);
            b1.h[0] = *(const bf16x8*)(wrow1 + cib + kA0);
            b1.h[1] = *(const bf16x8*)(wrow1 + cib + kA1);

            acc00 = __builtin_amdgcn_wmma_f32_16x16x32_bf16(false, a0.v, false, b0.v, (short)0, acc00, false, false);
            acc01 = __builtin_amdgcn_wmma_f32_16x16x32_bf16(false, a0.v, false, b1.v, (short)0, acc01, false, false);
            acc10 = __builtin_amdgcn_wmma_f32_16x16x32_bf16(false, a1.v, false, b0.v, (short)0, acc10, false, false);
            acc11 = __builtin_amdgcn_wmma_f32_16x16x32_bf16(false, a1.v, false, b1.v, (short)0, acc11, false, false);
        }
    }

    // Epilogue: bias + optional residual, store bf16 padded NHWC.
    const float bv0 = bias[cobase + ln];
    const float bv1 = bias[cobase + 16 + ln];
#pragma unroll
    for (int vr = 0; vr < 8; ++vr) {
        const int pm = half * 8 + vr;                      // D-layout pixel-in-tile
        const size_t ppA = (size_t)(row + 1) * PW + (c0 + pm + 1);
        const size_t ppB = ppA + 16;
        const size_t i00 = ppA * CHN + cobase + ln;
        const size_t i01 = i00 + 16;
        const size_t i10 = ppB * CHN + cobase + ln;
        const size_t i11 = i10 + 16;
        float v00 = acc00[vr] + bv0, v01 = acc01[vr] + bv1;
        float v10 = acc10[vr] + bv0, v11 = acc11[vr] + bv1;
        if (residual) {
            v00 += (float)residual[i00]; v01 += (float)residual[i01];
            v10 += (float)residual[i10]; v11 += (float)residual[i11];
        }
        out[i00] = (__bf16)v00; out[i01] = (__bf16)v01;
        out[i10] = (__bf16)v10; out[i11] = (__bf16)v11;
    }
}

// ---------------------------------------------------------------------------
// cnn6: 64 -> 1 channel. padded bf16 NHWC act -> f32 g[p] (halo => no checks)
// ---------------------------------------------------------------------------
__global__ __launch_bounds__(256) void conv6_kernel(
    const __bf16* __restrict__ in, const float* __restrict__ w6 /*[64][9]*/,
    const float* __restrict__ b6, float* __restrict__ g, int H)
{
    __shared__ float sw[64 * 9];
    for (int i = threadIdx.x; i < 576; i += 256) sw[i] = w6[i];
    __syncthreads();

    long long p = (long long)blockIdx.x * blockDim.x + threadIdx.x;
    if (p >= (long long)H * IMG_W) return;
    int r = (int)(p >> 6), c = (int)(p & 63);
    float acc = b6[0];
#pragma unroll
    for (int t = 0; t < 9; ++t) {
        const __bf16* arow = in + ((size_t)(r + t / 3) * PW + (c + t % 3)) * CHN;
#pragma unroll
        for (int ci = 0; ci < CHN; ++ci) acc += (float)arow[ci] * sw[ci * 9 + t];
    }
    g[p] = acc;
}

// ---------------------------------------------------------------------------
// Host-side orchestration
// ---------------------------------------------------------------------------
static inline int blks(long long n, int t) { return (int)((n + t - 1) / t); }

extern "C" void kernel_launch(void* const* d_in, const int* in_sizes, int n_in,
                              void* d_out, int out_size, void* d_ws, size_t ws_size,
                              hipStream_t stream) {
    const float*     x      = (const float*)d_in[0];
    const long long* ei     = (const long long*)d_in[1];
    const float*     gcn1_W = (const float*)d_in[2];
    const float*     gcn1_b = (const float*)d_in[3];
    const float*     gcn2_W = (const float*)d_in[4];
    const float*     gcn2_b = (const float*)d_in[5];
    const float*     cnn1_W = (const float*)d_in[6];
    const float*     cnn1_b = (const float*)d_in[7];
    const float*     cnn2_W = (const float*)d_in[8];
    const float*     cnn2_b = (const float*)d_in[9];
    const float*     cnn3_W = (const float*)d_in[10];
    const float*     cnn3_b = (const float*)d_in[11];
    const float*     cnn6_W = (const float*)d_in[12];
    const float*     cnn6_b = (const float*)d_in[13];
    const float*     lin2_W = (const float*)d_in[14];
    const float*     lin2_b = (const float*)d_in[15];

    const int       N = in_sizes[0] / 64;       // 16384 nodes / image height
    const long long E = in_sizes[1] / 2;        // 524288 edges
    const long long P = (long long)N * IMG_W;   // interior pixels
    const size_t    PPAD = (size_t)(N + 2) * PW;  // padded pixels per buffer

    // Workspace carve-out (256B aligned slices)
    char* ws = (char*)d_ws;
    size_t off = 0;
    auto take = [&](size_t bytes) -> char* {
        char* p = ws + off;
        off = (off + bytes + 255) & ~(size_t)255;
        return p;
    };
    float*  dis   = (float*)take((size_t)N * 4);           // deg -> D^-1/2 (in place)
    float*  xw1   = (float*)take((size_t)P * 4);
    float*  agg1  = (float*)take((size_t)P * 4);
    float*  h     = (float*)take((size_t)P * 4);           // GCN1 output (h_0)
    float*  g     = (float*)take((size_t)P * 4);           // cnn6 output
    float*  y     = (float*)take((size_t)P * 4);           // lin2 output
    float*  xw2   = (float*)take((size_t)N * 32 * 4);
    float*  agg2  = (float*)take((size_t)N * 32 * 4);
    __bf16* wbuf2 = (__bf16*)take((size_t)9 * 64 * 64 * 2);
    __bf16* wbuf3 = (__bf16*)take((size_t)9 * 64 * 64 * 2);
    __bf16* B0    = (__bf16*)take(PPAD * CHN * 2);         // ~138 MB each (padded)
    __bf16* B1    = (__bf16*)take(PPAD * CHN * 2);
    __bf16* B2    = (__bf16*)take(PPAD * CHN * 2);

    // ---- GCN layer 1 --------------------------------------------------------
    fill_zero_f32<<<blks(N, 256), 256, 0, stream>>>(dis, N);
    fill_zero_f32<<<blks(P, 256), 256, 0, stream>>>(agg1, P);
    fill_zero_f32<<<blks((long long)N * 32, 256), 256, 0, stream>>>(agg2, (long long)N * 32);

    degree_kernel<<<blks(E, 256), 256, 0, stream>>>(ei, E, dis);
    rsqrt_kernel<<<blks(N, 256), 256, 0, stream>>>(dis, N);
    gemm_small_kernel<<<blks((long long)N * 64, 256), 256, 0, stream>>>(
        x, nullptr, gcn1_W, nullptr, xw1, N, 64, 64);
    scatter_kernel<<<blks(E, 256), 256, 0, stream>>>(xw1, dis, ei, E, agg1, 64);
    gcn_finalize_kernel<<<blks(P, 256), 256, 0, stream>>>(h, agg1, xw1, dis, gcn1_b, N, 64);

    // ---- Conv weight conversion + halo zeroing -----------------------------
    weight_cvt_kernel<<<blks(64 * 64 * 9, 256), 256, 0, stream>>>(cnn2_W, wbuf2);
    weight_cvt_kernel<<<blks(64 * 64 * 9, 256), 256, 0, stream>>>(cnn3_W, wbuf3);
    const long long nPad = 132 + 2LL * N;
    pad_zero_kernel<<<blks(nPad, 256), 256, 0, stream>>>(B0, N);
    pad_zero_kernel<<<blks(nPad, 256), 256, 0, stream>>>(B1, N);
    pad_zero_kernel<<<blks(nPad, 256), 256, 0, stream>>>(B2, N);

    // ---- CNN stack (bf16 padded NHWC, WMMA for the 64->64 convs) -----------
    cnn1_kernel<<<blks(P, 256), 256, 0, stream>>>(h, cnn1_W, cnn1_b, B0, N);

    const int conv_grid = N / 2;            // 2 rows (128 px) per 8-wave block
    const size_t lds_bytes = (size_t)9 * 64 * 64 * 2;   // 72 KB weight stage
    // residual block 1: out_1 = conv3(conv2(out_0)) + out_0
    conv3x3_wmma_kernel<<<conv_grid, 256, lds_bytes, stream>>>(B0, wbuf2, cnn2_b, nullptr, B1, N);
    conv3x3_wmma_kernel<<<conv_grid, 256, lds_bytes, stream>>>(B1, wbuf3, cnn3_b, B0, B2, N);
    // residual block 2: out = conv3(conv2(out_1)) + out_1
    conv3x3_wmma_kernel<<<conv_grid, 256, lds_bytes, stream>>>(B2, wbuf2, cnn2_b, nullptr, B1, N);
    conv3x3_wmma_kernel<<<conv_grid, 256, lds_bytes, stream>>>(B1, wbuf3, cnn3_b, B2, B0, N);

    conv6_kernel<<<blks(P, 256), 256, 0, stream>>>(B0, cnn6_W, cnn6_b, g, N);

    // ---- lin2: y = (g + h_0) @ lin2_W + lin2_b -----------------------------
    gemm_small_kernel<<<blks((long long)N * 64, 256), 256, 0, stream>>>(
        g, h, lin2_W, lin2_b, y, N, 64, 64);

    // ---- GCN layer 2 -> d_out ----------------------------------------------
    gemm_small_kernel<<<blks((long long)N * 32, 256), 256, 0, stream>>>(
        y, nullptr, gcn2_W, nullptr, xw2, N, 64, 32);
    scatter_kernel<<<blks(E, 256), 256, 0, stream>>>(xw2, dis, ei, E, agg2, 32);
    gcn_finalize_kernel<<<blks((long long)N * 32, 256), 256, 0, stream>>>(
        (float*)d_out, agg2, xw2, dis, gcn2_b, N, 32);
}
